// Conv2dRepeat_39049842655588
// MI455X (gfx1250) — matrix operand order repaired and checked by reading, not compiled, not run on hardware
//
#include <hip/hip_runtime.h>

// ---------------- CDNA5 types ----------------
typedef __attribute__((ext_vector_type(16))) __bf16 v16bf;
typedef __attribute__((ext_vector_type(8)))  float  v8f;
typedef __attribute__((ext_vector_type(4)))  unsigned int u32x4;

// gcc-style vector-of-4-int, matching the async-to-LDS builtin prototype
typedef int v4i_gcc __attribute__((vector_size(4 * sizeof(int))));
typedef __attribute__((address_space(1))) v4i_gcc* gas_v4i;
typedef __attribute__((address_space(3))) v4i_gcc* las_v4i;

// ---------------- problem constants ----------------
#define CO    256
#define CI    256
#define HW    56
#define HW2   (HW*HW)          // 3136
#define NB    32
#define NPIX  (NB*HW2)         // 100352
#define KTAPS 9
#define KTILE 32               // bf16 WMMA K
#define KITERS (KTAPS*(CI/KTILE)) // 72
#define M_TILE 256             // all output channels per block
#define N_TILE 64              // pixels per block
#define LROW   40              // padded LDS row in ushorts (80 B, 16B aligned)

// async global->LDS path (CDNA5), guarded: falls back to reg pipeline
#if defined(__gfx1250__) && __has_builtin(__builtin_amdgcn_global_load_async_to_lds_b128)
#define HAVE_ASYNC_LDS 1
#endif

__device__ __forceinline__ unsigned short f2bf(float f) {
    unsigned int u = __float_as_uint(f);
    u += 0x7FFFu + ((u >> 16) & 1u);   // round to nearest even
    return (unsigned short)(u >> 16);
}

// -------------------------------------------------------------------------
// Stage 1: expand 64x64x3x3 base weight through swish-like activation into
// bf16, pre-swizzled: Wg[((tap*8 + cb)*256 + co)*32 + ci_in], ci = cb*32+ci_in
// -------------------------------------------------------------------------
__global__ __launch_bounds__(256)
void build_weight_kernel(const float* __restrict__ w,
                         const float* __restrict__ alphas,
                         const float* __restrict__ betas,
                         unsigned short* __restrict__ Wg) {
    int idx   = blockIdx.x * 256 + threadIdx.x;      // 0 .. 589823
    int ci_in = idx & 31;
    int co    = (idx >> 5) & 255;
    int tcb   = idx >> 13;                            // 0..71
    int cb    = tcb & 7;
    int t     = tcb >> 3;                             // tap 0..8
    int kh    = t / 3, kw = t - 3 * kh;
    int ci    = cb * 32 + ci_in;
    int p0 = co >> 6, i = co & 63;
    int p1 = ci >> 6, j = ci & 63;
    float wv = w[((i * 64 + j) * 3 + kh) * 3 + kw];
    float a  = alphas[p0 * 4 + p1];
    float b  = betas [p0 * 4 + p1];
    float val = wv * a / (1.0f + __expf(wv * b));
    Wg[idx] = f2bf(val);
}

// -------------------------------------------------------------------------
// Stage 2: implicit-GEMM conv via v_wmma_f32_16x16x32_bf16.
// Block: 256 threads (8 waves), tile 256 co x 64 px, double-buffered LDS,
// K loop of 72 x 32 (tap-major). Each wave: 32co x 64px = 8 WMMA / K-step.
// -------------------------------------------------------------------------
__global__ __launch_bounds__(256)
void conv_wmma_kernel(const float* __restrict__ x,
                      const unsigned short* __restrict__ Wg,
                      const float* __restrict__ bias,
                      float* __restrict__ out) {
    __shared__ unsigned short lA[2][M_TILE * LROW];   // 2 x 20 KB
    __shared__ unsigned short lB[2][N_TILE * LROW];   // 2 x  5 KB

    const int tid   = threadIdx.x;
    const int lane  = tid & 31;
    const int laneq = lane & 15;
    const int hi    = lane >> 4;          // 0: lanes 0-15, 1: lanes 16-31
    const int wv    = tid >> 5;           // wave 0..7 -> 32-co strip

    const int pix_base = blockIdx.x * N_TILE;         // 0..1567 blocks

    // ---- per-thread pixel decomposition for B staging (fixed all K iters)
    const int s_px = tid & 63;                    // pixel within tile
    const int s_cg = tid >> 6;                    // ci group 0..3 (8 ci each)
    const int sp   = pix_base + s_px;
    const int s_n  = sp / HW2;
    const int s_r  = sp - s_n * HW2;
    const int s_oh = s_r / HW;
    const int s_ow = s_r - s_oh * HW;
    const float* xpix = x + ((size_t)s_n * CI) * HW2 + s_oh * HW + s_ow;

    v8f acc[2][4];
#pragma unroll
    for (int i = 0; i < 2; ++i)
#pragma unroll
        for (int j = 0; j < 4; ++j) acc[i][j] = (v8f)0.0f;

    union Frag { u32x4 u[2]; v16bf v; };

    // staging registers
#if !defined(HAVE_ASYNC_LDS)
    u32x4 ra[4];                 // A tile: 16 KB / 256 thr = 64 B / thread
#endif
    float rb[8];                 // B tile: 8 ci values of one pixel

    // ---------------- staging helpers ----------------
#if defined(HAVE_ASYNC_LDS)
    auto asyncA = [&](int kk, int buf) {
        const u32x4* gp = (const u32x4*)(Wg + (size_t)kk * CO * KTILE);
#pragma unroll
        for (int s = 0; s < 4; ++s) {
            int g = tid + 256 * s;                    // 1024 16B chunks
            __builtin_amdgcn_global_load_async_to_lds_b128(
                (gas_v4i)(gp + g),
                (las_v4i)(lA[buf] + (g >> 2) * LROW + (g & 3) * 8),
                0, 0);
        }
    };
#else
    auto loadA = [&](int kk) {
        const u32x4* gp = (const u32x4*)(Wg + (size_t)kk * CO * KTILE);
#pragma unroll
        for (int s = 0; s < 4; ++s) ra[s] = gp[tid + 256 * s];
    };
    auto storeA = [&](int buf) {
#pragma unroll
        for (int s = 0; s < 4; ++s) {
            int g = tid + 256 * s;
            *(u32x4*)(lA[buf] + (g >> 2) * LROW + (g & 3) * 8) = ra[s];
        }
    };
#endif

    auto loadB = [&](int kk) {
        const int t  = kk >> 3;                       // tap 0..8
        const int cb = kk & 7;                        // ci block 0..7
        const int kh = t / 3, kw = t - 3 * kh;
        const int ih = s_oh + kh - 1;
        const int iw = s_ow + kw - 1;
        const bool valid = (ih >= 0) & (ih < HW) & (iw >= 0) & (iw < HW);
        const float* src = xpix + (size_t)(cb * 32 + s_cg * 8) * HW2
                                + (kh - 1) * HW + (kw - 1);
#pragma unroll
        for (int s = 0; s < 8; ++s)
            rb[s] = valid ? src[(size_t)s * HW2] : 0.0f;
    };
    auto storeB = [&](int buf) {
        unsigned int* drow = (unsigned int*)(lB[buf] + s_px * LROW + s_cg * 8);
#pragma unroll
        for (int s = 0; s < 4; ++s)
            drow[s] = (unsigned int)f2bf(rb[2 * s]) |
                      ((unsigned int)f2bf(rb[2 * s + 1]) << 16);
    };

    auto compute = [&](int buf) {
        Frag fa[2];
#pragma unroll
        for (int i = 0; i < 2; ++i) {
            // A 16x32 bf16: lanes0-15 v0-3=K0..7, v4-7=K16..23
            //               lanes16-31 v0-3=K8..15, v4-7=K24..31
            const u32x4* rp =
                (const u32x4*)(lA[buf] + (wv * 32 + i * 16 + laneq) * LROW);
            fa[i].u[0] = rp[hi];
            fa[i].u[1] = rp[hi + 2];
        }
#pragma unroll
        for (int j = 0; j < 4; ++j) {
            // B 32x16 bf16: lanes0-15 K0..7,K16..23, lanes16-31 K8..15,K24..31
            Frag fb;
            const u32x4* rp =
                (const u32x4*)(lB[buf] + (j * 16 + laneq) * LROW);
            fb.u[0] = rp[hi];
            fb.u[1] = rp[hi + 2];
#pragma unroll
            for (int i = 0; i < 2; ++i)
                acc[i][j] = __builtin_amdgcn_wmma_f32_16x16x32_bf16(
                    false, fa[i].v, false, fb.v,
                    (short)0, acc[i][j], false, false);
        }
    };

    // ---------------- prologue: stage tile 0 into buffer 0 ----------------
#if defined(HAVE_ASYNC_LDS)
    asyncA(0, 0);
#else
    loadA(0);
#endif
    loadB(0);
#if !defined(HAVE_ASYNC_LDS)
    storeA(0);
#endif
    storeB(0);
#if defined(HAVE_ASYNC_LDS)
    asm volatile("s_wait_asynccnt 0" ::: "memory");
#endif
    __syncthreads();

    // ---------------- pipelined main loop ----------------
    for (int kk = 0; kk < KITERS; ++kk) {
        const int cur  = kk & 1;
        const int nxt  = cur ^ 1;
        const bool more = (kk + 1) < KITERS;

        if (more) {                       // issue next tile's global loads
#if defined(HAVE_ASYNC_LDS)
            asyncA(kk + 1, nxt);
#else
            loadA(kk + 1);
#endif
            loadB(kk + 1);
        }

        compute(cur);                     // 8 WMMAs per wave

        if (more) {                       // publish next tile
#if !defined(HAVE_ASYNC_LDS)
            storeA(nxt);
#endif
            storeB(nxt);
#if defined(HAVE_ASYNC_LDS)
            asm volatile("s_wait_asynccnt 0" ::: "memory");
#endif
        }
        __syncthreads();
    }

    // ---------------- epilogue: C layout VGPR r -> M = r + 8*hi, N = laneq
#pragma unroll
    for (int j = 0; j < 4; ++j) {
        const int px = pix_base + j * 16 + laneq;
        const int n  = px / HW2;
        const int rm = px - n * HW2;
        const int oh = rm / HW;
        const int ow = rm - oh * HW;
        float* obase = out + ((size_t)n * CO) * HW2 + oh * HW + ow;
#pragma unroll
        for (int i = 0; i < 2; ++i) {
#pragma unroll
            for (int r = 0; r < 8; ++r) {
                const int co = wv * 32 + i * 16 + r + hi * 8;
                obase[(size_t)co * HW2] = acc[i][j][r] + bias[co];
            }
        }
    }
}

// -------------------------------------------------------------------------
extern "C" void kernel_launch(void* const* d_in, const int* in_sizes, int n_in,
                              void* d_out, int out_size, void* d_ws, size_t ws_size,
                              hipStream_t stream) {
    const float* x      = (const float*)d_in[0];
    const float* weight = (const float*)d_in[1];
    const float* alphas = (const float*)d_in[2];
    const float* betas  = (const float*)d_in[3];
    const float* bias   = (const float*)d_in[4];
    float* out = (float*)d_out;

    unsigned short* Wg = (unsigned short*)d_ws;   // 9*8*256*32 bf16 = 1.125 MB

    // Stage 1: expand weights (589824 elements)
    build_weight_kernel<<<(KTAPS * 8 * CO * KTILE) / 256, 256, 0, stream>>>(
        weight, alphas, betas, Wg);

    // Stage 2: implicit-GEMM conv, 1568 pixel-blocks of 64.
    conv_wmma_kernel<<<NPIX / N_TILE, 256, 0, stream>>>(x, Wg, bias, out);
}
